// GeometricSDEContiformer_77627238908460
// MI455X (gfx1250) — matrix operand order repaired and checked by reading, not compiled, not run on hardware
//
#include <hip/hip_runtime.h>
#include <math.h>

// ---------------- problem constants ----------------
#define B_    256
#define S_    512
#define H_    128
#define IN_   3
#define C_    5
#define NSUB_ 4
#define CAPV  409          // int(0.8*512)
#define EPSV  1e-8f

// WMMA fragment vector types (gfx1250, wave32)
typedef __attribute__((ext_vector_type(16))) _Float16 v16h;
typedef __attribute__((ext_vector_type(8)))  float    v8f;
typedef __attribute__((ext_vector_type(4)))  unsigned v4u;
typedef __attribute__((ext_vector_type(8)))  unsigned v8u;

// packed-weight buffer layout (dword offsets), per net:
//  L1: K=128 (t-row handled analytically), N=256 -> KT=4, NT=16 -> 64 frags
//  L2: K=256, N=128 -> KT=8, NT=8 -> 64 frags
//  L3: K=128, N=128 -> KT=4, NT=8 -> 32 frags
// one fragment = 32 lanes x 8 dwords = 256 dwords (1KB)
#define L1_OFF 0
#define L2_OFF 16384
#define L3_OFF 32768
#define NET_DW 40960       // dwords per net; total 2 nets = 81920 dw = 320KB
#define L12_DW 32768       // L1+L2 contiguous chunk per net (staged to LDS)

static __device__ inline v8f wmma_f16(v16h a, v16h b, v8f c) {
  // v_wmma_f32_16x16x32_f16  D = A*B + C   (f32 accumulate)
  return __builtin_amdgcn_wmma_f32_16x16x32_f16(false, a, false, b, (short)0, c,
                                                false, false);
}

// A-operand gather from row-major f16 activation tile [16][rowH] in LDS.
// Lane l: row m = l&15; K-base kb = (l<16)?0:8 (ISA 16-bit A layout).
// VGPRs 0..3 <- K = kt*32+kb+{0..7}, VGPRs 4..7 <- K = kt*32+kb+16+{0..7}
static __device__ inline v16h ldA(const _Float16* base, int rowH, int m, int kt, int kb) {
  const uint4* p = (const uint4*)(base + m * rowH + kt * 32 + kb);
  union { uint4 u[2]; v16h h; } f;
  f.u[0] = p[0];
  f.u[1] = p[2];   // +32 bytes
  return f.h;
}

// B-operand: pre-packed fragment, 8 dwords per lane, contiguous (32B).
static __device__ inline v16h ldB(const unsigned* fr) {
  const uint4* p = (const uint4*)fr;
  union { uint4 u[2]; v16h h; } f;
  f.u[0] = p[0];
  f.u[1] = p[1];
  return f.h;
}

// CDNA5 async copy: global -> LDS, 16B per lane, tracked by ASYNCcnt.
static __device__ inline void asyncLoadLDS128(unsigned ldsAddr, const void* g) {
  asm volatile("global_load_async_to_lds_b128 %0, %1, off"
               :: "v"(ldsAddr), "v"((unsigned long long)(size_t)g)
               : "memory");
}
static __device__ inline void waitAsync0() {
  asm volatile("s_wait_asynccnt 0x0" ::: "memory");
}

// ---------------- kernel arguments ----------------
struct SDEArgs {
  const float* ts;        // time_series (B,S,IN)
  const float* noise;     // (S-1, NSUB, B, H)
  const float* encW; const float* encB; const float* encG; const float* encBeta;
  const float* muB1; const float* muB2; const float* muB3;
  const float* sgB1; const float* sgB2; const float* sgB3;
  const float* minSig; const float* maxSig;
  const unsigned* packed; // 81920 dwords, f16 WMMA-B fragments
  const float* tmeans;    // (S)
  const int*   flags;     // (S-1)
  const float* w1last;    // [2][256] f32: t-row of w1 per net
  float* traj;            // d_out + B*C  -> (B,S,H)
  float* finalBuf;        // (B,H) mean over S
};

// ---------------- weight packing ----------------
__global__ __launch_bounds__(256) void pack_kernel(
    const float* muW1, const float* muW2, const float* muW3,
    const float* sgW1, const float* sgW2, const float* sgW3,
    unsigned* packed, float* w1last) {
  int bid = blockIdx.x;          // 0..319 : net*160 + fragment
  int t   = threadIdx.x;         // lane*8 + j
  int net = bid / 160;
  int rem = bid % 160;

  const float* W; int NT, layOff, srcLd, kt, nt;
  if (rem < 64)       { W = net ? sgW1 : muW1; NT = 16; layOff = net*NET_DW + L1_OFF; srcLd = 256; kt = rem/16;        nt = rem%16; }
  else if (rem < 128) { int fi = rem-64;  W = net ? sgW2 : muW2; NT = 8; layOff = net*NET_DW + L2_OFF; srcLd = 128; kt = fi/8; nt = fi%8; }
  else                { int fi = rem-128; W = net ? sgW3 : muW3; NT = 8; layOff = net*NET_DW + L3_OFF; srcLd = 128; kt = fi/8; nt = fi%8; }

  int lane = t >> 3, j = t & 7;
  // B-matrix 32x16 f16 layout: lanes 0-15 -> K 0..7,16..23 (col N=lane),
  // lanes 16-31 -> K 8..15,24..31 (col N=lane-16); VGPR j holds K pair kb+2j (j<4) / kb+16+2(j-4).
  int n  = nt * 16 + (lane & 15);
  int k0 = kt * 32 + ((lane < 16) ? 0 : 8) + ((j < 4) ? 2 * j : 16 + 2 * (j - 4));
  union { _Float16 h[2]; unsigned u; } pk;
  pk.h[0] = (_Float16)W[(k0    ) * srcLd + n];
  pk.h[1] = (_Float16)W[(k0 + 1) * srcLd + n];
  packed[layOff + ((kt * NT + nt) * 32 + lane) * 8 + j] = pk.u;

  if (bid == 0) w1last[t]       = muW1[128 * 256 + t];   // t-row of mu w1
  if (bid == 1) w1last[256 + t] = sgW1[128 * 256 + t];   // t-row of sg w1
}

// ---------------- t_means / flags ----------------
__global__ __launch_bounds__(512) void tstat_kernel(const float* times,
                                                    float* tmeans, int* flags) {
  __shared__ int   incs[S_];
  __shared__ float tms[S_];
  int s = threadIdx.x;
  float sum = 0.f; int inc = 0;
  for (int b = 0; b < B_; ++b) {
    float a = times[b * S_ + s];
    sum += a;
    if (s < S_ - 1) {
      float d = times[b * S_ + s + 1] - a;
      if (d > 1e-8f) inc = 1;
    }
  }
  float tm = sum * (1.0f / B_);
  tmeans[s] = tm; tms[s] = tm; incs[s] = inc;
  __syncthreads();
  if (s == 0) {
    int excl = 0;
    for (int i = 0; i < S_ - 1; ++i) {
      int f = (incs[i] && (excl < CAPV) && (tms[i + 1] > tms[i])) ? 1 : 0;
      flags[i] = f;
      excl += incs[i];
    }
  }
}

// encoder: Linear(3->128) + LayerNorm + ReLU + Softplus, for 16 rows, into LDS
static __device__ void computeEnc(const SDEArgs& A, int wg, int s, int tid,
                                  float* encL, float* mstat) {
  int rowm = tid >> 4;          // 0..15
  int fg   = (tid & 15) * 8;    // feature base
  const float* tsp = A.ts + (((size_t)(wg * 16 + rowm)) * S_ + s) * IN_;
  float t0 = tsp[0], t1 = tsp[1], t2 = tsp[2];
  for (int j = 0; j < 8; ++j) {
    int f = fg + j;
    encL[rowm * H_ + f] =
        t0 * A.encW[f] + t1 * A.encW[H_ + f] + t2 * A.encW[2 * H_ + f] + A.encB[f];
  }
  __syncthreads();
  if (tid < 16) {
    float s1 = 0.f, s2 = 0.f;
    for (int k = 0; k < H_; ++k) { float v = encL[tid * H_ + k]; s1 += v; s2 += v * v; }
    float mean = s1 * (1.0f / H_);
    float var  = s2 * (1.0f / H_) - mean * mean;
    mstat[tid * 2]     = mean;
    mstat[tid * 2 + 1] = rsqrtf(var + 1e-5f);
  }
  __syncthreads();
  for (int j = 0; j < 8; ++j) {
    int f = fg + j;
    float x = encL[rowm * H_ + f];
    x = (x - mstat[rowm * 2]) * mstat[rowm * 2 + 1] * A.encG[f] + A.encBeta[f];
    x = fmaxf(x, 0.f);
    x = (x > 20.f) ? x : log1pf(expf(x));           // softplus(relu)
    encL[rowm * H_ + f] = x;
  }
  __syncthreads();
}

// one 3-layer MLP; D-layout result for this wave's N-tile.
// L1/L2 B-fragments come from LDS (TDM-staged), L3 streams from L2 cache.
static __device__ void runNet(int net, const SDEArgs& A, const unsigned* wLds,
                              const _Float16* ysafeL, _Float16* h1L, _Float16* h2L,
                              float t, const float* b1, const float* b2,
                              const float* b3, int w, int l, float out[8]) {
  int ma = l & 15;
  int kb = (l < 16) ? 0 : 8;
  int md = (l < 16) ? 0 : 8;
  int na = w * 16 + (l & 15);
  int nb = na + 128;
  int nl = na;

  // ---- L1: y_safe(16x128) @ W1(128x256), this wave: N-tiles w and w+8
  const unsigned* base1 = wLds + net * L12_DW;          // L1 frags
  const unsigned* base2 = base1 + 16384;                // L2 frags
  const unsigned* g3 = A.packed + net * NET_DW + L3_OFF;
  __builtin_prefetch(g3 + (w * 32 + l) * 8, 0, 1);      // global_prefetch_b8
  v8f a1 = {}; v8f a2 = {};
  for (int kt = 0; kt < 4; ++kt) {
    v16h av = ldA(ysafeL, 128, ma, kt, kb);
    v16h b0 = ldB(base1 + ((kt * 16 + w    ) * 32 + l) * 8);
    v16h bb = ldB(base1 + ((kt * 16 + w + 8) * 32 + l) * 8);
    a1 = wmma_f16(av, b0, a1);
    a2 = wmma_f16(av, bb, a2);
  }
  const float* wl = A.w1last + net * 256;
  float ba  = b1[na] + t * wl[na];
  float bb2 = b1[nb] + t * wl[nb];
  for (int r = 0; r < 8; ++r) {
    h1L[(md + r) * 256 + na] = (_Float16)tanhf(a1[r] + ba);
    h1L[(md + r) * 256 + nb] = (_Float16)tanhf(a2[r] + bb2);
  }
  __syncthreads();

  // ---- L2: h1(16x256) @ W2(256x128), B from LDS
  v8f c2 = {};
  for (int kt = 0; kt < 8; ++kt) {
    v16h av = ldA(h1L, 256, ma, kt, kb);
    v16h bv = ldB(base2 + ((kt * 8 + w) * 32 + l) * 8);
    c2 = wmma_f16(av, bv, c2);
  }
  for (int r = 0; r < 8; ++r)
    h2L[(md + r) * 128 + nl] = (_Float16)tanhf(c2[r] + b2[nl]);
  __syncthreads();

  // ---- L3: h2(16x128) @ W3(128x128), B streamed from L2-resident buffer
  v8f c3 = {};
  for (int kt = 0; kt < 4; ++kt) {
    v16h av = ldA(h2L, 128, ma, kt, kb);
    v16h bv = ldB(g3 + ((kt * 8 + w) * 32 + l) * 8);
    c3 = wmma_f16(av, bv, c3);
  }
  for (int r = 0; r < 8; ++r) out[r] = c3[r] + b3[nl];
}

// ---------------- main persistent SDE scan ----------------
__global__ __launch_bounds__(256) void sde_main(SDEArgs A) {
  __shared__ unsigned wL[2 * L12_DW];      // 256KB: L1+L2 packed, both nets
  __shared__ _Float16 ysafeL[16 * 128];    // 4KB
  __shared__ _Float16 h1L[16 * 256];       // 8KB
  __shared__ _Float16 h2L[16 * 128];       // 4KB
  __shared__ float    encL[16 * 128];      // 8KB
  __shared__ float    noiseL[16 * 128];    // 8KB, async-prefetched dW tile
  __shared__ float    mstat[16 * 2];

  int tid = threadIdx.x, wg = blockIdx.x;
  int w = tid >> 5, l = tid & 31;
  int md = (l < 16) ? 0 : 8;
  int nl = w * 16 + (l & 15);

  // -------- TDM: stage L1+L2 fragments of both nets into LDS --------
  // 2D tensor in dwords: tile 32768 x 2, row stride NET_DW (skips L3 chunk).
  if (tid < 32) {   // wave 0 issues the descriptor (TDM ignores EXEC)
    unsigned long long ga = (unsigned long long)(size_t)A.packed;
    unsigned ldsb = (unsigned)(size_t)(&wL[0]);          // LDS byte offset
    v4u g0; v8u g1;
    g0[0] = 1u;                                          // count=1, user mode
    g0[1] = ldsb;                                        // lds_addr (bytes)
    g0[2] = (unsigned)(ga & 0xffffffffu);                // global_addr[31:0]
    g0[3] = (unsigned)((ga >> 32) & 0x1ffffffu)          // global_addr[56:32]
            | 0x80000000u;                               // type=2 ("image")
    g1[0] = 2u << 16;                                    // data_size=4B
    g1[1] = (unsigned)(L12_DW & 0xffff) << 16;           // tensor_dim0 lo16
    g1[2] = ((unsigned)(L12_DW >> 16) & 0xffff)          // tensor_dim0 hi16
            | (2u << 16);                                // tensor_dim1 lo16 = 2
    g1[3] = (unsigned)(L12_DW & 0xffff) << 16;           // tile_dim0 = 32768
    g1[4] = 2u;                                          // tile_dim1 = 2
    g1[5] = (unsigned)NET_DW;                            // dim0_stride[31:0]
    g1[6] = 0u;                                          // stride hi / dim1_stride lo
    g1[7] = 0u;
    asm volatile("tensor_load_to_lds %0, %1" :: "s"(g0), "s"(g1) : "memory");
    __builtin_amdgcn_s_wait_tensorcnt(0);
  }
  __syncthreads();

  float smin = fabsf(A.minSig[0]);
  float smax = fabsf(A.maxSig[0]);
  unsigned noiseBase = (unsigned)(size_t)(&noiseL[0]);

  // state0 = pos(encoded[:,0])
  computeEnc(A, wg, 0, tid, encL, mstat);
  float y[8], acc[8];
  for (int r = 0; r < 8; ++r) {
    float v = fminf(fmaxf(fabsf(encL[(md + r) * H_ + nl]), EPSV), 10.f);
    y[r] = v; acc[r] = v;
    A.traj[((size_t)(wg * 16 + md + r) * S_ + 0) * H_ + nl] = v;
  }

  for (int step = 0; step < S_ - 1; ++step) {
    __syncthreads();   // protect encL reads of previous iteration
    float t0 = A.tmeans[step], t1 = A.tmeans[step + 1];
    float h = fmaxf((t1 - t0) * (1.0f / NSUB_), 1e-12f);
    float sqh = sqrtf(h);

    computeEnc(A, wg, step + 1, tid, encL, mstat);   // enc_i for this step

    for (int sub = 0; sub < NSUB_; ++sub) {
      float t = t0 + h * (float)sub;
      float ys[8];
      for (int r = 0; r < 8; ++r) {
        ys[r] = fminf(fmaxf(fabsf(y[r]), EPSV), 10.f);
        ysafeL[(md + r) * H_ + nl] = (_Float16)ys[r];
      }
      __syncthreads();

      // async-prefetch this substep's 8KB dW tile; overlaps both MLPs
      {
        const char* gnp = (const char*)(A.noise +
            (((size_t)(step * NSUB_ + sub)) * B_ + wg * 16) * H_);
        asyncLoadLDS128(noiseBase + tid * 16,        gnp + tid * 16);
        asyncLoadLDS128(noiseBase + 4096 + tid * 16, gnp + 4096 + tid * 16);
      }

      float mu[8], sg[8];
      runNet(0, A, wL, ysafeL, h1L, h2L, t, A.muB1, A.muB2, A.muB3, w, l, mu);
      runNet(1, A, wL, ysafeL, h1L, h2L, t, A.sgB1, A.sgB2, A.sgB3, w, l, sg);

      waitAsync0();        // own wave's async copies done
      __syncthreads();     // all waves' copies visible

      for (int r = 0; r < 8; ++r) {
        float m_ = fminf(fmaxf(mu[r], -5.f), 5.f);
        float s_ = fminf(fmaxf(sg[r], smin), smax);
        float dW = noiseL[(md + r) * H_ + nl];
        y[r] = y[r] + m_ * ys[r] * h + s_ * ys[r] * sqh * dW;
      }
    }

    int flag = A.flags[step];
    for (int r = 0; r < 8; ++r) {
      float cand = flag ? y[r] : encL[(md + r) * H_ + nl];
      float v = fminf(fmaxf(fabsf(cand), EPSV), 10.f);
      y[r] = v; acc[r] += v;
      A.traj[((size_t)(wg * 16 + md + r) * S_ + (step + 1)) * H_ + nl] = v;
    }
  }

  for (int r = 0; r < 8; ++r)
    A.finalBuf[(wg * 16 + md + r) * H_ + nl] = acc[r] * (1.0f / S_);
}

// ---------------- classifier ----------------
__global__ __launch_bounds__(64) void cls_kernel(const float* finalBuf,
                                                 const float* w1, const float* b1,
                                                 const float* w2, const float* b2,
                                                 float* logits) {
  __shared__ float hid[64];
  int b = blockIdx.x, j = threadIdx.x;
  const float* f = finalBuf + b * H_;
  float s = 0.f;
  for (int k = 0; k < H_; ++k) s += f[k] * w1[k * 64 + j];
  hid[j] = fmaxf(s + b1[j], 0.f);
  __syncthreads();
  if (j < C_) {
    float s2 = 0.f;
    for (int k = 0; k < 64; ++k) s2 += hid[k] * w2[k * C_ + j];
    logits[b * C_ + j] = s2 + b2[j];
  }
}

// ---------------- launch ----------------
extern "C" void kernel_launch(void* const* d_in, const int* in_sizes, int n_in,
                              void* d_out, int out_size, void* d_ws, size_t ws_size,
                              hipStream_t stream) {
  (void)in_sizes; (void)n_in; (void)out_size; (void)ws_size;
  // setup_inputs() insertion order, dicts flattened in insertion order:
  const float* ts      = (const float*)d_in[0];
  const float* times   = (const float*)d_in[1];
  const float* noise   = (const float*)d_in[2];
  const float* encW    = (const float*)d_in[3];
  const float* encB    = (const float*)d_in[4];
  const float* encG    = (const float*)d_in[5];
  const float* encBeta = (const float*)d_in[6];
  const float* muW1 = (const float*)d_in[7];  const float* muB1 = (const float*)d_in[8];
  const float* muW2 = (const float*)d_in[9];  const float* muB2 = (const float*)d_in[10];
  const float* muW3 = (const float*)d_in[11]; const float* muB3 = (const float*)d_in[12];
  const float* sgW1 = (const float*)d_in[13]; const float* sgB1 = (const float*)d_in[14];
  const float* sgW2 = (const float*)d_in[15]; const float* sgB2 = (const float*)d_in[16];
  const float* sgW3 = (const float*)d_in[17]; const float* sgB3 = (const float*)d_in[18];
  const float* clsW1 = (const float*)d_in[19]; const float* clsB1 = (const float*)d_in[20];
  const float* clsW2 = (const float*)d_in[21]; const float* clsB2 = (const float*)d_in[22];
  const float* minSig = (const float*)d_in[23];
  const float* maxSig = (const float*)d_in[24];

  unsigned char* ws = (unsigned char*)d_ws;
  unsigned* packed  = (unsigned*)ws;                         // 327,680 B
  float*    tmeans  = (float*)(ws + 327680);                 // 2,048 B
  int*      flags   = (int*)  (ws + 327680 + 2048);          // 2,048 B
  float*    w1last  = (float*)(ws + 327680 + 4096);          // 2,048 B
  float*    finalBf = (float*)(ws + 327680 + 6144);          // 131,072 B

  float* logits = (float*)d_out;
  float* traj   = logits + B_ * C_;

  pack_kernel<<<320, 256, 0, stream>>>(muW1, muW2, muW3, sgW1, sgW2, sgW3,
                                       packed, w1last);
  tstat_kernel<<<1, 512, 0, stream>>>(times, tmeans, flags);

  SDEArgs A;
  A.ts = ts; A.noise = noise;
  A.encW = encW; A.encB = encB; A.encG = encG; A.encBeta = encBeta;
  A.muB1 = muB1; A.muB2 = muB2; A.muB3 = muB3;
  A.sgB1 = sgB1; A.sgB2 = sgB2; A.sgB3 = sgB3;
  A.minSig = minSig; A.maxSig = maxSig;
  A.packed = packed; A.tmeans = tmeans; A.flags = flags; A.w1last = w1last;
  A.traj = traj; A.finalBuf = finalBf;

  sde_main<<<B_ / 16, 256, 0, stream>>>(A);
  cls_kernel<<<B_, 64, 0, stream>>>(finalBf, clsW1, clsB1, clsW2, clsB2, logits);
}
